// RawSDP_60078002537020
// MI455X (gfx1250) — matrix operand order repaired and checked by reading, not compiled, not run on hardware
//
#include <hip/hip_runtime.h>

typedef unsigned short u16;
typedef unsigned int   u32;
typedef __attribute__((ext_vector_type(16))) __bf16 v16bf;
typedef __attribute__((ext_vector_type(8)))  float  v8f;

#define Bn 4
#define Hn 16
#define Sn 2048
#define Dn 128
#define BM 128          // query rows per workgroup (8 waves x 16)
#define BN 64           // keys per iteration
#define KS 136          // K LDS row stride in halves (272B, 16B-aligned, padded)
#define VS 80           // V^T LDS row stride in halves (160B, 16B-aligned, padded)
#define PS 80           // P LDS row stride in halves

__device__ __forceinline__ u16 f2bf(float x) {
  u32 u = __float_as_uint(x);
  u32 r = (u + 0x7FFFu + ((u >> 16) & 1u)) >> 16;   // round-to-nearest-even
  return (u16)r;
}

union BV { uint4 q[2]; v16bf v; };

__global__ __launch_bounds__(256)
void sdp_flash_bf16_wmma(const float* __restrict__ Q, const float* __restrict__ K,
                         const float* __restrict__ V, float* __restrict__ O) {
  __shared__ __align__(16) u16 sK[BN * KS];        // K block, row-major  [key][d]
  __shared__ __align__(16) u16 sVt[Dn * VS];       // V block, transposed [d][key]
  __shared__ __align__(16) u16 sP[8 * 16 * PS];    // per-wave P scratch  [m][k]

  const int tid  = threadIdx.x;
  const int w    = tid >> 5;
  const int lane = tid & 31;
  const int nl   = lane & 15;      // column / row-in-half index
  const int hi   = lane >> 4;      // half-wave select

  const int qb = blockIdx.x;
  const int bh = blockIdx.y;
  const size_t base = (size_t)bh * Sn * Dn;
  const float* Qb = Q + base;
  const float* Kb = K + base;
  const float* Vb = V + base;
  float*       Ob = O + base;

  const int qrow0 = qb * BM + w * 16;
  const float scale = 0.08838834764831845f;   // 1/sqrt(128) == (d^-1/4)^2

  // staging coords: 4 threads per key row, 32 d-elements each
  const int srow = tid >> 2;
  const int sc0  = (tid & 3) * 32;

  // ---- Preload Q tile (16 x 128) in WMMA A layout, scaled, bf16 ----
  v16bf qv[4];
  {
    const float* qrow = Qb + (size_t)(qrow0 + nl) * Dn;
    const int kb = hi * 8;
    #pragma unroll
    for (int c = 0; c < 4; ++c) {
      const float4* p0 = (const float4*)(qrow + c * 32 + kb);
      const float4* p1 = (const float4*)(qrow + c * 32 + kb + 16);
      float4 f0 = p0[0], f1 = p0[1], f2 = p1[0], f3 = p1[1];
      float fs[16] = {f0.x,f0.y,f0.z,f0.w, f1.x,f1.y,f1.z,f1.w,
                      f2.x,f2.y,f2.z,f2.w, f3.x,f3.y,f3.z,f3.w};
      u32 pk[8];
      #pragma unroll
      for (int j = 0; j < 8; ++j)
        pk[j] = (u32)f2bf(fs[2*j] * scale) | ((u32)f2bf(fs[2*j+1] * scale) << 16);
      BV bv;
      bv.q[0] = make_uint4(pk[0], pk[1], pk[2], pk[3]);
      bv.q[1] = make_uint4(pk[4], pk[5], pk[6], pk[7]);
      qv[c] = bv.v;
    }
  }

  v8f o[8];
  #pragma unroll
  for (int nt = 0; nt < 8; ++nt) {
    v8f z = {0.f,0.f,0.f,0.f,0.f,0.f,0.f,0.f};
    o[nt] = z;
  }
  float mrow[8], lrow[8];
  #pragma unroll
  for (int r = 0; r < 8; ++r) { mrow[r] = -3.0e38f; lrow[r] = 0.f; }

  u16* pw = sP + w * 16 * PS;

  for (int n0 = 0; n0 < Sn; n0 += BN) {
    // ---- stage K (row-major) and V (transposed) to LDS as bf16 ----
    {
      const float2* kp = (const float2*)(Kb + (size_t)(n0 + srow) * Dn + sc0);
      const float2* vp = (const float2*)(Vb + (size_t)(n0 + srow) * Dn + sc0);
      #pragma unroll
      for (int j = 0; j < 16; ++j) {
        float2 kk = kp[j];
        *(u32*)&sK[srow * KS + sc0 + 2*j] =
            (u32)f2bf(kk.x) | ((u32)f2bf(kk.y) << 16);
      }
      #pragma unroll
      for (int j = 0; j < 16; ++j) {
        float2 vv = vp[j];
        sVt[(sc0 + 2*j    ) * VS + srow] = f2bf(vv.x);
        sVt[(sc0 + 2*j + 1) * VS + srow] = f2bf(vv.y);
      }
    }
    __syncthreads();

    if (n0 + BN < Sn) {   // prefetch next block -> global_prefetch_b8
      __builtin_prefetch(Kb + (size_t)(n0 + BN + srow) * Dn + sc0, 0, 0);
      __builtin_prefetch(Vb + (size_t)(n0 + BN + srow) * Dn + sc0, 0, 0);
    }

    // ---- S = Q K^T : 4 key-tiles x 4 d-chunks of wmma bf16 ----
    v8f s[4];
    #pragma unroll
    for (int t = 0; t < 4; ++t) {
      v8f acc = {0.f,0.f,0.f,0.f,0.f,0.f,0.f,0.f};
      #pragma unroll
      for (int c = 0; c < 4; ++c) {
        BV bv;
        const uint4* bp = (const uint4*)&sK[(t*16 + nl) * KS + c*32 + hi*8];
        bv.q[0] = bp[0]; bv.q[1] = bp[2];
        acc = __builtin_amdgcn_wmma_f32_16x16x32_bf16(
            false, qv[c], false, bv.v, (short)0, acc, false, false);
      }
      s[t] = acc;
    }

    // ---- fp32 online softmax (row = r + 8*hi, col across lanes in half) ----
    float alpha[8];
    #pragma unroll
    for (int r = 0; r < 8; ++r) {
      float v = fmaxf(fmaxf(s[0][r], s[1][r]), fmaxf(s[2][r], s[3][r]));
      v = fmaxf(v, __shfl_xor(v, 1, 32));
      v = fmaxf(v, __shfl_xor(v, 2, 32));
      v = fmaxf(v, __shfl_xor(v, 4, 32));
      v = fmaxf(v, __shfl_xor(v, 8, 32));
      float mn = fmaxf(mrow[r], v);
      alpha[r] = __expf(mrow[r] - mn);
      mrow[r] = mn;
    }
    #pragma unroll
    for (int t = 0; t < 4; ++t) {
      #pragma unroll
      for (int r = 0; r < 8; ++r)
        s[t][r] = __expf(s[t][r] - mrow[r]);
    }
    #pragma unroll
    for (int r = 0; r < 8; ++r) {
      float ps = s[0][r] + s[1][r] + s[2][r] + s[3][r];
      ps += __shfl_xor(ps, 1, 32);
      ps += __shfl_xor(ps, 2, 32);
      ps += __shfl_xor(ps, 4, 32);
      ps += __shfl_xor(ps, 8, 32);
      lrow[r] = lrow[r] * alpha[r] + ps;
    }
    #pragma unroll
    for (int nt = 0; nt < 8; ++nt) {
      #pragma unroll
      for (int r = 0; r < 8; ++r)
        o[nt][r] *= alpha[r];
    }

    // ---- P -> per-wave LDS scratch (bf16, [16 x 64]) ----
    #pragma unroll
    for (int t = 0; t < 4; ++t) {
      #pragma unroll
      for (int r = 0; r < 8; ++r)
        pw[(r + 8*hi) * PS + t*16 + nl] = f2bf(s[t][r]);
    }
    // same-wave LDS RAW: in-order DS, no barrier needed

    // ---- O += P V : 2 key-chunks x 8 d-tiles of wmma bf16 ----
    #pragma unroll
    for (int kc = 0; kc < 2; ++kc) {
      BV av;
      const uint4* ap0 = (const uint4*)&pw[nl * PS + kc*32 + hi*8];
      const uint4* ap1 = (const uint4*)&pw[nl * PS + kc*32 + hi*8 + 16];
      av.q[0] = ap0[0]; av.q[1] = ap1[0];
      #pragma unroll
      for (int nt = 0; nt < 8; ++nt) {
        BV bv;
        const uint4* bp = (const uint4*)&sVt[(nt*16 + nl) * VS + kc*32 + hi*8];
        bv.q[0] = bp[0]; bv.q[1] = bp[2];
        o[nt] = __builtin_amdgcn_wmma_f32_16x16x32_bf16(
            false, av.v, false, bv.v, (short)0, o[nt], false, false);
      }
    }
    __syncthreads();   // protect sK/sVt before next staging
  }

  // ---- epilogue: normalize and store fp32 ----
  float inv[8];
  #pragma unroll
  for (int r = 0; r < 8; ++r) inv[r] = 1.0f / lrow[r];
  #pragma unroll
  for (int nt = 0; nt < 8; ++nt) {
    #pragma unroll
    for (int r = 0; r < 8; ++r) {
      Ob[(size_t)(qrow0 + r + 8*hi) * Dn + nt*16 + nl] = o[nt][r] * inv[r];
    }
  }
}

extern "C" void kernel_launch(void* const* d_in, const int* in_sizes, int n_in,
                              void* d_out, int out_size, void* d_ws, size_t ws_size,
                              hipStream_t stream) {
  const float* Q = (const float*)d_in[0];
  const float* K = (const float*)d_in[1];
  const float* V = (const float*)d_in[2];
  float* O = (float*)d_out;
  dim3 grid(Sn / BM, Bn * Hn);   // 16 query blocks x 64 (b,h) heads
  sdp_flash_bf16_wmma<<<grid, dim3(256), 0, stream>>>(Q, K, V, O);
}
